// SpikingLayer_60799557042852
// MI455X (gfx1250) — matrix operand order repaired and checked by reading, not compiled, not run on hardware
//
#include <hip/hip_runtime.h>

// ---------------------------------------------------------------------------
// SpikingLayer (LIF over a linear layer) for MI455X / gfx1250, wave32.
//
//   current = x @ W^T + b           (4096 x 1024 x 1024, f32)
//   20x:  v = v*0.5 + current; spike = (v-1 > 0); v -= spike
//   outputs: spikes[B,T,F], v_mems[B,T,F]  (concatenated flat, f32)
//
// Store-bound problem (671 MB out @ 23.3 TB/s ~ 29us floor), so GEMM + all 20
// LIF steps are fused in one kernel. GEMM runs on V_WMMA_F32_16X16X32_BF16
// with a hi/lo bf16 split of both operands (3 WMMAs/k-step) for ~fp32
// accuracy at bf16 matrix-engine rates; inputs (20 MB) live in L2 (192 MB).
// ---------------------------------------------------------------------------

typedef __bf16 v8bf  __attribute__((ext_vector_type(8)));
typedef __bf16 v16bf __attribute__((ext_vector_type(16)));
typedef float  v8f   __attribute__((ext_vector_type(8)));

#define B_DIM   4096
#define F_IN    1024
#define F_OUT   1024
#define T_STEPS 20

// ---- one-time f32 -> (bf16 hi, bf16 lo) split into workspace --------------
__global__ __launch_bounds__(256) void split_bf16_kernel(
    const float* __restrict__ in, __bf16* __restrict__ hi,
    __bf16* __restrict__ lo, int n) {
  int i = blockIdx.x * blockDim.x + threadIdx.x;
  if (i < n) {
    float f  = in[i];
    __bf16 h = (__bf16)f;
    hi[i] = h;
    lo[i] = (__bf16)(f - (float)h);
  }
}

// ---- fused GEMM (split-bf16 WMMA) + LIF scan ------------------------------
// WG = 256 threads = 8 waves, wave grid 4(M) x 2(N).
// Wave tile: 32x32 via 2x2 V_WMMA_F32_16X16X32_BF16 tiles, f32 accum.
// WG tile: 128(M) x 64(N); grid = (1024/64, 4096/128) = (16, 32).
__global__ __launch_bounds__(256) void gemm_lif_kernel(
    const __bf16* __restrict__ xh, const __bf16* __restrict__ xl,
    const __bf16* __restrict__ wh, const __bf16* __restrict__ wl,
    const float* __restrict__ bias, float* __restrict__ out) {
  const int lane  = threadIdx.x & 31;
  const int wave  = threadIdx.x >> 5;
  const int wm    = wave & 3;
  const int wn    = wave >> 2;
  const int m_base = blockIdx.y * 128 + wm * 32;
  const int n_base = blockIdx.x * 64  + wn * 32;

  const int lmod  = lane & 15;   // row (A) / col (B,C)
  const int lhalf = lane >> 4;   // K-half selector (A,B) / M+8 selector (C)

  v8f acc[2][2] = {};

  // A (16x32) k order chosen here (contiguous, matches B): lane holds row M=lmod,
  //   elems 0..15 -> K = k0 + lhalf*16 .. +15     (contiguous 32B)
  // B (32x16) k order chosen here (contiguous, matches A): lane holds col N=lmod,
  //   elems 0..15 -> K = k0 + lhalf*16 .. +15     (contiguous 32B)
  const size_t arow0 = (size_t)(m_base + lmod)      * F_IN;
  const size_t arow1 = (size_t)(m_base + 16 + lmod) * F_IN;
  const size_t brow0 = (size_t)(n_base + lmod)      * F_IN;
  const size_t brow1 = (size_t)(n_base + 16 + lmod) * F_IN;
  const int aoff = lhalf * 16;
  const int boff = lhalf * 16;

  for (int k0 = 0; k0 < F_IN; k0 += 32) {
    union Frag { v16bf v; v8bf h[2]; };
    Frag a_hi[2], a_lo[2], b_hi[2], b_lo[2];

    {
      const __bf16* p = xh + arow0 + k0 + aoff;
      a_hi[0].h[0] = *(const v8bf*)p;        a_hi[0].h[1] = *(const v8bf*)(p + 8);
      const __bf16* q = xl + arow0 + k0 + aoff;
      a_lo[0].h[0] = *(const v8bf*)q;        a_lo[0].h[1] = *(const v8bf*)(q + 8);
      p = xh + arow1 + k0 + aoff;
      a_hi[1].h[0] = *(const v8bf*)p;        a_hi[1].h[1] = *(const v8bf*)(p + 8);
      q = xl + arow1 + k0 + aoff;
      a_lo[1].h[0] = *(const v8bf*)q;        a_lo[1].h[1] = *(const v8bf*)(q + 8);
    }
    {
      const __bf16* p = wh + brow0 + k0 + boff;
      b_hi[0].h[0] = *(const v8bf*)p;        b_hi[0].h[1] = *(const v8bf*)(p + 8);
      const __bf16* q = wl + brow0 + k0 + boff;
      b_lo[0].h[0] = *(const v8bf*)q;        b_lo[0].h[1] = *(const v8bf*)(q + 8);
      p = wh + brow1 + k0 + boff;
      b_hi[1].h[0] = *(const v8bf*)p;        b_hi[1].h[1] = *(const v8bf*)(p + 8);
      q = wl + brow1 + k0 + boff;
      b_lo[1].h[0] = *(const v8bf*)q;        b_lo[1].h[1] = *(const v8bf*)(q + 8);
    }

#pragma unroll
    for (int mt = 0; mt < 2; ++mt) {
#pragma unroll
      for (int nt = 0; nt < 2; ++nt) {
        // x*W ~= xh*wh + xh*wl + xl*wh  (lo*lo term ~2^-18, dropped)
        acc[mt][nt] = __builtin_amdgcn_wmma_f32_16x16x32_bf16(
            false, a_hi[mt].v, false, b_hi[nt].v, (short)0, acc[mt][nt], false, false);
        acc[mt][nt] = __builtin_amdgcn_wmma_f32_16x16x32_bf16(
            false, a_hi[mt].v, false, b_lo[nt].v, (short)0, acc[mt][nt], false, false);
        acc[mt][nt] = __builtin_amdgcn_wmma_f32_16x16x32_bf16(
            false, a_lo[mt].v, false, b_hi[nt].v, (short)0, acc[mt][nt], false, false);
      }
    }
  }

  // C/D layout: VGPR r -> (M = lhalf*8 + r, N = lmod) within each 16x16 tile.
  const size_t ROW_STRIDE = (size_t)T_STEPS * F_OUT;   // per-M stride in out
  float* __restrict__ spikes = out;
  float* __restrict__ vmems  = out + (size_t)B_DIM * T_STEPS * F_OUT;

#pragma unroll
  for (int mt = 0; mt < 2; ++mt) {
#pragma unroll
    for (int nt = 0; nt < 2; ++nt) {
      const int n   = n_base + nt * 16 + lmod;
      const float bv = bias[n];
      v8f cur = acc[mt][nt];
#pragma unroll
      for (int r = 0; r < 8; ++r) cur[r] += bv;

      const int m0 = m_base + mt * 16 + lhalf * 8;
      v8f v = {};
      for (int t = 0; t < T_STEPS; ++t) {
        const size_t obase = (size_t)t * F_OUT + n;
#pragma unroll
        for (int r = 0; r < 8; ++r) {
          float vv = v[r] * 0.5f + cur[r];          // leak (TAU=0.5) + integrate
          float sp = (vv > 1.0f) ? 1.0f : 0.0f;     // heaviside(v - V_TH)
          vv -= sp;                                 // subtract-reset (V_TH=1)
          const size_t off = (size_t)(m0 + r) * ROW_STRIDE + obase;
          spikes[off] = sp;
          vmems[off]  = vv;
          v[r] = vv;
        }
      }
    }
  }
}

extern "C" void kernel_launch(void* const* d_in, const int* in_sizes, int n_in,
                              void* d_out, int out_size, void* d_ws, size_t ws_size,
                              hipStream_t stream) {
  const float* x = (const float*)d_in[0];   // [4096, 1024]
  const float* W = (const float*)d_in[1];   // [1024, 1024]
  const float* b = (const float*)d_in[2];   // [1024]
  float* out = (float*)d_out;               // spikes ++ v_mems, f32

  // workspace layout: xh(8MB) xl(8MB) wh(2MB) wl(2MB) = 20 MB
  __bf16* xh = (__bf16*)d_ws;
  __bf16* xl = xh + (size_t)B_DIM * F_IN;
  __bf16* wh = xl + (size_t)B_DIM * F_IN;
  __bf16* wl = wh + (size_t)F_OUT * F_IN;

  const int nx = B_DIM * F_IN;    // 4194304
  const int nw = F_OUT * F_IN;    // 1048576
  split_bf16_kernel<<<(nx + 255) / 256, 256, 0, stream>>>(x, xh, xl, nx);
  split_bf16_kernel<<<(nw + 255) / 256, 256, 0, stream>>>(W, wh, wl, nw);

  dim3 grid(F_OUT / 64, B_DIM / 128);   // (16, 32)
  gemm_lif_kernel<<<grid, 256, 0, stream>>>(xh, xl, wh, wl, b, out);
}
